// MultiHeadAttentionBlock_5299989644052
// MI455X (gfx1250) — matrix operand order repaired and checked by reading, not compiled, not run on hardware
//
#include <hip/hip_runtime.h>
#include <hip/hip_bf16.h>

// ---------- types ----------
typedef __bf16 bf16_t;
typedef __attribute__((ext_vector_type(16))) __bf16 v16bf;
typedef __attribute__((ext_vector_type(8)))  __bf16 v8bf;
typedef __attribute__((ext_vector_type(8)))  float  v8f;
typedef __attribute__((ext_vector_type(4)))  float  v4f;
typedef __attribute__((ext_vector_type(4)))  unsigned int u32x4;
typedef __attribute__((ext_vector_type(8)))  unsigned int u32x8;

#define BB 4
#define CC 2048
#define DD 1024
#define HH 8
#define KK 64
#define VV 128

// K-chunk staged per TDM transfer (elements) and padded LDS row stride.
// 64 bf16 = 128 B = 32 DWORDs per row -> pad_interval code 4 (32 DW),
// pad_amount code 3 (4 DW = 8 bf16) -> LDS row stride 72 elements (144 B).
#define KCH   64
#define RS    72

// ---------- scalar bf16 helpers ----------
__device__ __forceinline__ bf16_t f2bf(float f) {
    unsigned int u = __builtin_bit_cast(unsigned int, f);
    unsigned int r = u + 0x7fffu + ((u >> 16) & 1u);
    return __builtin_bit_cast(bf16_t, (unsigned short)(r >> 16));
}
__device__ __forceinline__ unsigned short f2bf_bits(float f) {
    unsigned int u = __builtin_bit_cast(unsigned int, f);
    unsigned int r = u + 0x7fffu + ((u >> 16) & 1u);
    return (unsigned short)(r >> 16);
}
__device__ __forceinline__ float bf2f(bf16_t h) {
    unsigned int u = ((unsigned int)__builtin_bit_cast(unsigned short, h)) << 16;
    return __builtin_bit_cast(float, u);
}
__device__ __forceinline__ float bf2f_bits(unsigned int bits16) {
    unsigned int u = bits16 << 16;
    return __builtin_bit_cast(float, u);
}
__device__ __forceinline__ bf16_t to_bf(float f)  { return f2bf(f); }
__device__ __forceinline__ bf16_t to_bf(bf16_t h) { return h; }

// ---------- WMMA fragment loads ----------
// A (16x32 bf16 row-major): lane holds row M=lane&15; lanes 0-15: k=[0,8)+[16,24),
// lanes 16-31: k=[8,16)+[24,32). caller passes &A[row*lda + k0 + ((lane>>4)<<3)]
__device__ __forceinline__ v16bf load_a_frag(const bf16_t* p) {
    v8bf lo = *(const v8bf*)(p);
    v8bf hi = *(const v8bf*)(p + 16);
    return __builtin_shufflevector(lo, hi, 0,1,2,3,4,5,6,7,8,9,10,11,12,13,14,15);
}
// B (32x16 bf16) from a [N x K] row-major panel: lane holds column N=lane&15,
// lanes 0-15 k=[0,8)+[16,24), lanes 16-31 k=[8,16)+[24,32): 2x16B.
__device__ __forceinline__ v16bf load_b_frag(const bf16_t* p) {
    v8bf lo = *(const v8bf*)(p);
    v8bf hi = *(const v8bf*)(p + 16);
    return __builtin_shufflevector(lo, hi, 0,1,2,3,4,5,6,7,8,9,10,11,12,13,14,15);
}

// ---------- Tensor Data Mover: stage a [rows x KCH] bf16 tile into LDS ----------
// D# per CDNA5 ISA. 2D tensor, data_size=2B, LDS padding
// of 4 DWORDs every 32 DWORDs -> 72-element padded rows. Issued by one wave,
// tracked by TENSORcnt.
__device__ __forceinline__ void tdm_load_tile(unsigned lds_off_bytes,
                                              const bf16_t* gsrc,
                                              unsigned rows, unsigned ldb_elems) {
    unsigned long long ga = (unsigned long long)(size_t)gsrc;
    u32x4 g0;
    g0[0] = 1u;                                   // count=1 (valid user descriptor)
    g0[1] = lds_off_bytes;                        // lds_addr
    g0[2] = (unsigned)ga;                         // global_addr[31:0]
    g0[3] = (unsigned)((ga >> 32) & 0x01FFFFFFu)  // global_addr[56:32]
          | (2u << 30);                           // type=2 ("image")
    u32x8 g1;
    g1[0] = (1u << 16)                            // data_size = 1 -> 2 bytes
          | (1u << 20)                            // pad_enable
          | (4u << 22)                            // pad_interval: 32 DWORDs
          | (3u << 25);                           // pad_amount: 4 DWORDs
    g1[1] = ((unsigned)KCH & 0xffffu) << 16;      // tensor_dim0[15:0] @ bit48
    g1[2] = ((unsigned)KCH >> 16) | (rows << 16); // tensor_dim0[31:16] | tensor_dim1[15:0]
    g1[3] = (rows >> 16) | ((unsigned)KCH << 16); // tensor_dim1[31:16] | tile_dim0
    g1[4] = rows & 0xffffu;                       // tile_dim1 (tile_dim2 = 0)
    g1[5] = ldb_elems;                            // tensor_dim0_stride[31:0]
    g1[6] = 0u;
    g1[7] = 0u;
    asm volatile("tensor_load_to_lds %0, %1" :: "s"(g0), "s"(g1) : "memory");
}

// ---------- elementwise f32 -> bf16 convert ----------
__global__ void cvt_f32_bf16(const float* __restrict__ in, bf16_t* __restrict__ out, int n4) {
    int i = blockIdx.x * blockDim.x + threadIdx.x;
    if (i >= n4) return;
    v4f v = *(const v4f*)(in + (size_t)i * 4);
    bf16_t* o = out + (size_t)i * 4;
    o[0] = f2bf(v.x); o[1] = f2bf(v.y); o[2] = f2bf(v.z); o[3] = f2bf(v.w);
}

// ---------- LDS-tiled transpose (+convert), batched via blockIdx.z ----------
template<typename TIn>
__global__ void transpose_to_bf16(const TIn* __restrict__ in, bf16_t* __restrict__ out,
                                  int rows, int cols) {
    __shared__ bf16_t tile[32][33];
    size_t boff = (size_t)blockIdx.z * rows * cols;
    in += boff; out += boff;
    int c = blockIdx.x * 32 + threadIdx.x;
    #pragma unroll
    for (int i = 0; i < 4; ++i) {
        int r = blockIdx.y * 32 + threadIdx.y + i * 8;
        tile[threadIdx.y + i * 8][threadIdx.x] = to_bf(in[(size_t)r * cols + c]);
    }
    __syncthreads();
    int oc = blockIdx.y * 32 + threadIdx.x;
    #pragma unroll
    for (int i = 0; i < 4; ++i) {
        int orow = blockIdx.x * 32 + threadIdx.y + i * 8;
        out[(size_t)orow * rows + oc] = tile[threadIdx.x][threadIdx.y + i * 8];
    }
}

// ---------- wave compute step: batched B loads then WMMA chain w/ reuse_a ------
template<int NT>
__device__ __forceinline__ void wmma_chain(v16bf a, const bf16_t* bbuf, int kk,
                                           v8f (&acc)[NT]) {
    v16bf bfrag[NT];
    #pragma unroll
    for (int t = 0; t < NT; ++t)
        bfrag[t] = load_b_frag(bbuf + (size_t)(t * 16) * RS + kk);
    // identical back-to-back WMMAs sharing A: legal to set reuse_a after the first
    acc[0] = __builtin_amdgcn_wmma_f32_16x16x32_bf16(
        false, a, false, bfrag[0], (short)0, acc[0], false, false);
    #pragma unroll
    for (int t = 1; t < NT; ++t)
        acc[t] = __builtin_amdgcn_wmma_f32_16x16x32_bf16(
            false, a, false, bfrag[t], (short)0, acc[t], true, false);
}

// ---------- batched NT GEMM with TDM-staged, double-buffered B panel ----------
// C[m,n] = sum_k A[m,k]*Bt[n,k]. 8 waves; wave tile 16 x (NT*16); block 128 x NT*16.
template<int NT>
__global__ void gemm_nt_tdm(const bf16_t* __restrict__ A, int lda, size_t strideA,
                            const bf16_t* __restrict__ Bt, int ldb, size_t strideB,
                            bf16_t* __restrict__ Cm, int ldc, size_t strideC,
                            int kdim) {
    extern __shared__ bf16_t smem[];
    constexpr int NR = NT * 16;
    constexpr unsigned BUFE = (unsigned)NR * RS;   // elements per LDS buffer
    A  += (size_t)blockIdx.z * strideA;
    Bt += (size_t)blockIdx.z * strideB;
    Cm += (size_t)blockIdx.z * strideC;
    int lane = threadIdx.x & 31, w = threadIdx.x >> 5;
    int m0 = blockIdx.x * 128 + w * 16;
    int n0 = blockIdx.y * NR;
    const bf16_t* ap = A + (size_t)(m0 + (lane & 15)) * lda + ((lane >> 4) << 3);
    const bf16_t* gB = Bt + (size_t)n0 * ldb;
    unsigned lds0 = (unsigned)(size_t)smem;        // LDS byte offset (low 32 bits)
    bool issuer = threadIdx.x < 32;
    int nch = kdim / KCH;
    if (issuer) tdm_load_tile(lds0, gB, NR, (unsigned)ldb);
    v8f z = {}; v8f acc[NT];
    #pragma unroll
    for (int t = 0; t < NT; ++t) acc[t] = z;
    for (int c = 0; c < nch; ++c) {
        if (issuer) __builtin_amdgcn_s_wait_tensorcnt(0);
        __syncthreads();
        if (issuer && (c + 1) < nch)
            tdm_load_tile(lds0 + (unsigned)(((c + 1) & 1) * BUFE * 2),
                          gB + (size_t)(c + 1) * KCH, NR, (unsigned)ldb);
        const bf16_t* bbuf = smem + (size_t)(c & 1) * BUFE
                           + (size_t)(lane & 15) * RS + ((lane >> 4) << 3);
        int kg = c * KCH;
        #pragma unroll
        for (int kk = 0; kk < KCH; kk += 32) {
            __builtin_prefetch(ap + kg + kk + KCH, 0, 0);
            v16bf a = load_a_frag(ap + kg + kk);
            wmma_chain<NT>(a, bbuf, kk, acc);
        }
        __syncthreads();
    }
    int rbase = m0 + ((lane >> 4) << 3);
    int cbase = n0 + (lane & 15);
    #pragma unroll
    for (int t = 0; t < NT; ++t)
        #pragma unroll
        for (int rr = 0; rr < 8; ++rr)
            Cm[(size_t)(rbase + rr) * ldc + cbase + t * 16] = f2bf(acc[t][rr]);
}

// ---------- online column softmax over q (axis=1 of [BH, C, C]) ----------
// one u32 (two adjacent bf16 columns) per thread; coalesced strided passes.
__global__ void softmax_col(bf16_t* __restrict__ S) {
    int g = blockIdx.y;
    int c2 = blockIdx.x * 256 + threadIdx.x;        // column-pair index
    unsigned* base = (unsigned*)(S + (size_t)g * CC * CC) + c2;
    float m0 = -3.0e38f, m1 = -3.0e38f, s0 = 0.f, s1 = 0.f;
    for (int q = 0; q < CC; ++q) {
        unsigned u = base[(size_t)q * (CC / 2)];
        float v0 = bf2f_bits(u & 0xffffu) * 0.125f;   // 1/sqrt(64)
        float v1 = bf2f_bits(u >> 16)     * 0.125f;
        float nm0 = fmaxf(m0, v0), nm1 = fmaxf(m1, v1);
        s0 = s0 * __expf(m0 - nm0) + __expf(v0 - nm0);
        s1 = s1 * __expf(m1 - nm1) + __expf(v1 - nm1);
        m0 = nm0; m1 = nm1;
    }
    float i0 = 1.0f / s0, i1 = 1.0f / s1;
    for (int q = 0; q < CC; ++q) {
        unsigned u = base[(size_t)q * (CC / 2)];
        float v0 = bf2f_bits(u & 0xffffu) * 0.125f;
        float v1 = bf2f_bits(u >> 16)     * 0.125f;
        unsigned lo = f2bf_bits(__expf(v0 - m0) * i0);
        unsigned hi = f2bf_bits(__expf(v1 - m1) * i1);
        base[(size_t)q * (CC / 2)] = lo | (hi << 16);
    }
}

// ---------- AV GEMM: permuted A rows, full N=128 per wave, TDM-staged V panel ----
// out[b,q2,h2,d] = sum_c att[b, q2>>8, ((q2&255)<<3)+h2, c] * v[b,h2,c,d]
__global__ void av_gemm(const bf16_t* __restrict__ att, const bf16_t* __restrict__ Vt,
                        float* __restrict__ out) {
    extern __shared__ bf16_t smem[];
    constexpr int NR = VV;                          // 128 rows of Vt
    constexpr unsigned BUFE = (unsigned)NR * RS;
    int g = blockIdx.z, b = g >> 3, h2 = g & 7;
    int lane = threadIdx.x & 31, w = threadIdx.x >> 5;
    int m0 = blockIdx.x * 128 + w * 16;             // q2 tile
    int q2l = m0 + (lane & 15);
    int hx = q2l >> 8;
    int ar = ((q2l & 255) << 3) + h2;
    const bf16_t* ap = att + ((size_t)((b << 3) + hx) * CC + ar) * CC + ((lane >> 4) << 3);
    const bf16_t* gB = Vt + (size_t)g * VV * CC;
    unsigned lds0 = (unsigned)(size_t)smem;
    bool issuer = threadIdx.x < 32;
    int nch = CC / KCH;                             // 32
    if (issuer) tdm_load_tile(lds0, gB, NR, CC);
    v8f z = {}; v8f acc[8];
    #pragma unroll
    for (int t = 0; t < 8; ++t) acc[t] = z;
    for (int c = 0; c < nch; ++c) {
        if (issuer) __builtin_amdgcn_s_wait_tensorcnt(0);
        __syncthreads();
        if (issuer && (c + 1) < nch)
            tdm_load_tile(lds0 + (unsigned)(((c + 1) & 1) * BUFE * 2),
                          gB + (size_t)(c + 1) * KCH, NR, CC);
        const bf16_t* bbuf = smem + (size_t)(c & 1) * BUFE
                           + (size_t)(lane & 15) * RS + ((lane >> 4) << 3);
        int kg = c * KCH;
        #pragma unroll
        for (int kk = 0; kk < KCH; kk += 32) {
            __builtin_prefetch(ap + kg + kk + KCH, 0, 0);
            v16bf a = load_a_frag(ap + kg + kk);
            wmma_chain<8>(a, bbuf, kk, acc);
        }
        __syncthreads();
    }
    int rbase = m0 + ((lane >> 4) << 3);
    #pragma unroll
    for (int t = 0; t < 8; ++t)
        #pragma unroll
        for (int rr = 0; rr < 8; ++rr) {
            int q2 = rbase + rr;
            out[((size_t)b * CC + q2) * (HH * VV) + (h2 << 7) + t * 16 + (lane & 15)]
                = acc[t][rr];
        }
}

extern "C" void kernel_launch(void* const* d_in, const int* in_sizes, int n_in,
                              void* d_out, int out_size, void* d_ws, size_t ws_size,
                              hipStream_t stream) {
    (void)in_sizes; (void)n_in; (void)out_size; (void)ws_size;
    const float* x  = (const float*)d_in[0];   // [4,2048,1024]
    const float* Mq = (const float*)d_in[1];   // [1024,512]
    const float* Mk = (const float*)d_in[2];   // [1024,512]
    const float* Mv = (const float*)d_in[3];   // [1024,1024]
    float* out = (float*)d_out;                // [4,2048,1024]

    char* w = (char*)d_ws;
    size_t o = 0;
    bf16_t* xb  = (bf16_t*)(w + o); o += (size_t)BB*CC*DD    * 2;
    bf16_t* mqT = (bf16_t*)(w + o); o += (size_t)512*DD      * 2;
    bf16_t* mkT = (bf16_t*)(w + o); o += (size_t)512*DD      * 2;
    bf16_t* mvT = (bf16_t*)(w + o); o += (size_t)1024*DD     * 2;
    bf16_t* Qb  = (bf16_t*)(w + o); o += (size_t)BB*CC*512   * 2;
    bf16_t* Kb  = (bf16_t*)(w + o); o += (size_t)BB*CC*512   * 2;
    bf16_t* Vb  = (bf16_t*)(w + o); o += (size_t)BB*CC*1024  * 2;
    bf16_t* Vt  = (bf16_t*)(w + o); o += (size_t)BB*HH*VV*CC * 2;
    bf16_t* S   = (bf16_t*)(w + o); o += (size_t)BB*HH*CC*CC * 2;

    dim3 t256(256), t32x8(32, 8);
    const unsigned lds4 = 2u * 64 * RS * 2;    // double-buffered 64-row panel
    const unsigned lds8 = 2u * 128 * RS * 2;   // double-buffered 128-row panel

    // 1) x -> bf16
    cvt_f32_bf16<<<dim3((BB*CC*DD/4 + 255)/256), t256, 0, stream>>>(x, xb, BB*CC*DD/4);

    // 2) weights: transpose + convert
    transpose_to_bf16<float><<<dim3(512/32, DD/32, 1),  t32x8, 0, stream>>>(Mq, mqT, DD, 512);
    transpose_to_bf16<float><<<dim3(512/32, DD/32, 1),  t32x8, 0, stream>>>(Mk, mkT, DD, 512);
    transpose_to_bf16<float><<<dim3(1024/32, DD/32, 1), t32x8, 0, stream>>>(Mv, mvT, DD, 1024);

    // 3) projections (NT GEMM vs transposed weights, TDM-staged B)
    gemm_nt_tdm<4><<<dim3(BB*CC/128, 512/64, 1),  t256, lds4, stream>>>(
        xb, DD, 0, mqT, DD, 0, Qb, 512, 0, DD);
    gemm_nt_tdm<4><<<dim3(BB*CC/128, 512/64, 1),  t256, lds4, stream>>>(
        xb, DD, 0, mkT, DD, 0, Kb, 512, 0, DD);
    gemm_nt_tdm<4><<<dim3(BB*CC/128, 1024/64, 1), t256, lds4, stream>>>(
        xb, DD, 0, mvT, DD, 0, Vb, 1024, 0, DD);

    // 4) scores: per (b,h): S = Q_bh @ K_bh^T (group stride C*K)
    gemm_nt_tdm<4><<<dim3(CC/128, CC/64, BB*HH), t256, lds4, stream>>>(
        Qb, KK, (size_t)CC*KK, Kb, KK, (size_t)CC*KK, S, CC, (size_t)CC*CC, KK);

    // 5) online softmax over q, in place (includes 1/sqrt(64))
    softmax_col<<<dim3(CC/512, BB*HH), t256, 0, stream>>>(S);

    // 6) V: batched transpose [BH][2048][128] -> [BH][128][2048]
    transpose_to_bf16<bf16_t><<<dim3(VV/32, CC/32, BB*HH), t32x8, 0, stream>>>(Vb, Vt, CC, VV);

    // 7) out = att(permuted rows) @ V, one pass over att
    av_gemm<<<dim3(CC/128, 1, BB*HH), t256, lds8, stream>>>(S, Vt, out);
}
